// GNNClusterBridge_56349970923677
// MI455X (gfx1250) — matrix operand order repaired and checked by reading, not compiled, not run on hardware
//
#include <hip/hip_runtime.h>
#include <stdint.h>

#define FEAT 128

typedef __attribute__((ext_vector_type(16))) __bf16 v16bf;
typedef __attribute__((ext_vector_type(8)))  float  v8f;

// native fp32 -> bf16 (RNE); lets clang use the packed v_cvt path instead of
// manual bit-twiddling (3 VALU ops/element)
static __device__ __forceinline__ __bf16 f2bf(float f) { return (__bf16)f; }

__global__ void fill0_kernel(float* __restrict__ p, int n) {
  int i = blockIdx.x * blockDim.x + threadIdx.x;
  if (i < n) p[i] = 0.0f;
}

__global__ void deg_kernel(const int* __restrict__ dst, const float* __restrict__ ew,
                           float* __restrict__ deg, int n_edges) {
  int e = blockIdx.x * blockDim.x + threadIdx.x;
  if (e < n_edges) atomicAdd(&deg[dst[e]], ew[e]);
}

__global__ void norm_kernel(const float* __restrict__ deg, float* __restrict__ dis,
                            float* __restrict__ selfs, int n) {
  int i = blockIdx.x * blockDim.x + threadIdx.x;
  if (i < n) {
    float d = deg[i] + 1.0f;           // self-loop weight 1
    float r = rsqrtf(d);
    dis[i] = r;
    selfs[i] = r * r;                  // 1/deg
  }
}

__global__ void edge_norm_kernel(const int* __restrict__ src, const int* __restrict__ dst,
                                 const float* __restrict__ ew, const float* __restrict__ dis,
                                 float* __restrict__ enorm, int n_edges) {
  int e = blockIdx.x * blockDim.x + threadIdx.x;
  if (e < n_edges) enorm[e] = dis[src[e]] * dis[dst[e]] * ew[e];
}

// H[n_rows, NOUT] = X[n_rows, 128] @ W[128, NOUT]   (bf16 WMMA, f32 accum)
// If selfs != nullptr, additionally writes AGG = H * selfs[row] in the epilogue
// (fuses away a full 2x25.6MB init_self pass per layer).
template <int NOUT>
__global__ __launch_bounds__(256) void gemm_wmma_kernel(
    const float* __restrict__ X, const float* __restrict__ W,
    float* __restrict__ H, const float* __restrict__ selfs,
    float* __restrict__ AGG, int n_rows) {
  // W transposed into LDS as bf16: Wt[n][k]
  __shared__ __bf16 Wt[NOUT * FEAT];
  const int tid = threadIdx.x;
  for (int idx = tid; idx < FEAT * NOUT; idx += 256) {
    int k = idx / NOUT;
    int n = idx % NOUT;
    Wt[n * FEAT + k] = f2bf(W[idx]);
  }
  __syncthreads();

  const int wave = tid >> 5;
  const int lane = tid & 31;
  const int half = lane >> 4;       // which 16-lane group
  const int lmod = lane & 15;
  const long m_base = (long)blockIdx.x * 128 + (long)wave * 16;

  long arow = m_base + lmod;        // A-fragment row for this lane (both halves share M)
  if (arow >= n_rows) arow = n_rows - 1;   // clamp (tail rows masked at store)
  const float* xrow = X + arow * FEAT;

  constexpr int NT = NOUT / 16;
  v8f acc[NT] = {};

#pragma unroll
  for (int k0 = 0; k0 < FEAT; k0 += 32) {
    // A 16x32 bf16 fragment: lane holds K = k0+16h .. k0+16h+15 (contiguous)
    const float4* p0 = (const float4*)(xrow + k0 + 16 * half);
    const float4* p1 = (const float4*)(xrow + k0 + 16 * half + 8);
    float4 t0 = p0[0], t1 = p0[1], t2 = p1[0], t3 = p1[1];
    float lo[8] = {t0.x, t0.y, t0.z, t0.w, t1.x, t1.y, t1.z, t1.w};
    float hi[8] = {t2.x, t2.y, t2.z, t2.w, t3.x, t3.y, t3.z, t3.w};
    v16bf a;
#pragma unroll
    for (int i = 0; i < 8; ++i) { a[i] = f2bf(lo[i]); a[8 + i] = f2bf(hi[i]); }

#pragma unroll
    for (int t = 0; t < NT; ++t) {
      // B 32x16 fragment: lane holds col N=t*16+lmod, K = k0+16h .. k0+16h+15 (contig in Wt)
      const v16bf b = *(const v16bf*)(&Wt[(t * 16 + lmod) * FEAT + k0 + 16 * half]);
      acc[t] = __builtin_amdgcn_wmma_f32_16x16x32_bf16(
          /*neg_a=*/false, a, /*neg_b=*/false, b,
          /*c_mod=*/(short)0, acc[t], /*reuse_a=*/false, /*reuse_b=*/false);
    }
  }

  // C/D layout: VGPR j -> row j + 8*half, col = lane%16
  float sc[8];
  if (selfs != nullptr) {
#pragma unroll
    for (int j = 0; j < 8; ++j) {
      long row = m_base + j + 8 * half;
      sc[j] = (row < n_rows) ? selfs[row] : 0.0f;
    }
  }
#pragma unroll
  for (int t = 0; t < NT; ++t) {
#pragma unroll
    for (int j = 0; j < 8; ++j) {
      long row = m_base + j + 8 * half;
      if (row < n_rows) {
        float v = acc[t][j];
        long off = row * NOUT + t * 16 + lmod;
        H[off] = v;
        if (selfs != nullptr) AGG[off] = v * sc[j];
      }
    }
  }
}

// one wave per edge: lane covers 4 features; float4 gather + 4 atomic f32 adds
__global__ void scatter_kernel(const float* __restrict__ H, const int* __restrict__ src,
                               const int* __restrict__ dst, const float* __restrict__ enorm,
                               float* __restrict__ AGG, int n_edges) {
  long g = (long)blockIdx.x * blockDim.x + threadIdx.x;
  int e = (int)(g >> 5);
  int lane = threadIdx.x & 31;
  if (e >= n_edges) return;
  int s = src[e];
  int d = dst[e];
  float w = enorm[e];
  float4 v = ((const float4*)(H + (long)s * FEAT))[lane];
  float* outp = AGG + (long)d * FEAT + lane * 4;
  atomicAdd(outp + 0, v.x * w);
  atomicAdd(outp + 1, v.y * w);
  atomicAdd(outp + 2, v.z * w);
  atomicAdd(outp + 3, v.w * w);
}

// in-place: AGG = relu(AGG + bias[f])
__global__ void finalize_kernel(float* __restrict__ AGG, const float* __restrict__ bias,
                                int n_rows) {
  long i = (long)blockIdx.x * blockDim.x + threadIdx.x;
  if (i < (long)n_rows * FEAT) {
    float v = AGG[i] + bias[(int)(i & 127)];
    AGG[i] = v > 0.0f ? v : 0.0f;
  }
}

// one wave per row, 64 cols (2 per lane), in-place softmax of (logits + bc)
__global__ void softmax_kernel(float* __restrict__ out, const float* __restrict__ bc,
                               int n_rows) {
  long g = (long)blockIdx.x * blockDim.x + threadIdx.x;
  int row = (int)(g >> 5);
  int lane = threadIdx.x & 31;
  if (row >= n_rows) return;
  float* p = out + (long)row * 64;
  float v0 = p[lane] + bc[lane];
  float v1 = p[lane + 32] + bc[lane + 32];
  float m = fmaxf(v0, v1);
#pragma unroll
  for (int off = 16; off > 0; off >>= 1) m = fmaxf(m, __shfl_xor(m, off, 32));
  float e0 = __expf(v0 - m);
  float e1 = __expf(v1 - m);
  float s = e0 + e1;
#pragma unroll
  for (int off = 16; off > 0; off >>= 1) s += __shfl_xor(s, off, 32);
  float inv = 1.0f / s;
  p[lane] = e0 * inv;
  p[lane + 32] = e1 * inv;
}

extern "C" void kernel_launch(void* const* d_in, const int* in_sizes, int n_in,
                              void* d_out, int out_size, void* d_ws, size_t ws_size,
                              hipStream_t stream) {
  const float* x  = (const float*)d_in[0];
  const int*   ei = (const int*)d_in[1];
  const float* ew = (const float*)d_in[2];
  const float* W1 = (const float*)d_in[3];
  const float* b1 = (const float*)d_in[4];
  const float* W2 = (const float*)d_in[5];
  const float* b2 = (const float*)d_in[6];
  const float* W3 = (const float*)d_in[7];
  const float* b3 = (const float*)d_in[8];
  const float* Wc = (const float*)d_in[9];
  const float* bc = (const float*)d_in[10];

  const int N = in_sizes[0] / FEAT;
  const int E = in_sizes[2];
  const int* src = ei;       // edge_index[0]
  const int* dst = ei + E;   // edge_index[1]

  float* ws = (float*)d_ws;
  size_t o = 0;
  auto take = [&](size_t nfloat) {
    float* p = ws + o;
    o += (nfloat + 63) & ~size_t(63);
    return p;
  };
  float* deg   = take(N);
  float* dis   = take(N);
  float* selfs = take(N);
  float* enorm = take(E);
  float* F0    = take((size_t)N * FEAT);   // GEMM output h = xW
  float* F1    = take((size_t)N * FEAT);   // aggregation / layer output

  const int T = 256;
  const int nB  = (N + T - 1) / T;
  const int eB  = (E + T - 1) / T;
  const int gB  = (N + 127) / 128;                       // GEMM: 128 rows / block
  const int nfB = (int)(((long)N * FEAT + T - 1) / T);
  const int scB = (int)(((long)E * 32 + T - 1) / T);     // wave per edge
  const int smB = (int)(((long)N * 32 + T - 1) / T);     // wave per row

  // graph normalization, computed once and reused across layers
  fill0_kernel<<<nB, T, 0, stream>>>(deg, N);
  deg_kernel<<<eB, T, 0, stream>>>(dst, ew, deg, E);
  norm_kernel<<<nB, T, 0, stream>>>(deg, dis, selfs, N);
  edge_norm_kernel<<<eB, T, 0, stream>>>(src, dst, ew, dis, enorm, E);

  // layer 1: GEMM epilogue seeds AGG = h * self_scale (fused init_self)
  gemm_wmma_kernel<FEAT><<<gB, T, 0, stream>>>(x, W1, F0, selfs, F1, N);
  scatter_kernel<<<scB, T, 0, stream>>>(F0, src, dst, enorm, F1, E);
  finalize_kernel<<<nfB, T, 0, stream>>>(F1, b1, N);
  // layer 2
  gemm_wmma_kernel<FEAT><<<gB, T, 0, stream>>>(F1, W2, F0, selfs, F1, N);
  scatter_kernel<<<scB, T, 0, stream>>>(F0, src, dst, enorm, F1, E);
  finalize_kernel<<<nfB, T, 0, stream>>>(F1, b2, N);
  // layer 3
  gemm_wmma_kernel<FEAT><<<gB, T, 0, stream>>>(F1, W3, F0, selfs, F1, N);
  scatter_kernel<<<scB, T, 0, stream>>>(F0, src, dst, enorm, F1, E);
  finalize_kernel<<<nfB, T, 0, stream>>>(F1, b3, N);

  // cluster head: logits straight into d_out, then in-place softmax (+bc)
  gemm_wmma_kernel<64><<<gB, T, 0, stream>>>(F1, Wc, (float*)d_out, nullptr, nullptr, N);
  softmax_kernel<<<smB, T, 0, stream>>>((float*)d_out, bc, N);
}
